// Attention_29755533427251
// MI455X (gfx1250) — matrix operand order repaired and checked by reading, not compiled, not run on hardware
//
#include <hip/hip_runtime.h>
#include <hip/hip_bf16.h>

// ---------------------------------------------------------------------------
// Sparse talking-heads attention for MI455X (gfx1250), bf16 WMMA pipeline.
// Round 2: add CDNA5 async-to-LDS weight staging (ASYNCcnt) in the output GEMM.
// ---------------------------------------------------------------------------

#define B_    2
#define N_    2048
#define DIM_  1024
#define H_    16
#define DH_   64
#define MEM_  16
#define J_    2064      // MEM_ + N_
#define JK_   2112      // K rows padded to 66*32
#define JP_   2080      // j padded to 65*32 for attn2 / V^T
#define TOPK_ 64
#define NEGMAX (-3.402823466e38f)

typedef __attribute__((ext_vector_type(16))) __bf16 bfrag_t;
typedef __attribute__((ext_vector_type(8)))  float  v8f;

struct alignas(16) U128 { unsigned int w[4]; };
union FragU { bfrag_t v; U128 q[2]; };

__device__ __forceinline__ unsigned short f2bf(float f) {
  unsigned u = __float_as_uint(f);
  u += 0x7fffu + ((u >> 16) & 1u);          // round-to-nearest-even
  return (unsigned short)(u >> 16);
}
__device__ __forceinline__ float bf2f(unsigned short h) {
  return __uint_as_float(((unsigned)h) << 16);
}
__device__ __forceinline__ v8f vzero() {
  v8f z = {0.f,0.f,0.f,0.f,0.f,0.f,0.f,0.f};
  return z;
}
__device__ __forceinline__ v8f wmma_bf16(bfrag_t a, bfrag_t b, v8f c) {
  return __builtin_amdgcn_wmma_f32_16x16x32_bf16(false, a, false, b, (short)0, c,
                                                 false, false);
}

// A fragment: 16(M) x 32(K) bf16, row-major source (K contiguous, stride lda).
// Lanes 0-15: M=lane, K = k0+0..15 contiguous; lanes 16-31: K = k0+16..31.
__device__ __forceinline__ bfrag_t load_afrag(const unsigned short* base, int lda,
                                              int m0, int k0, int lane) {
  int row = m0 + (lane & 15);
  int kk  = k0 + ((lane >> 4) << 4);
  const unsigned short* p = base + (size_t)row * lda + kk;
  FragU f;
  f.q[0] = *(const U128*)(p);
  f.q[1] = *(const U128*)(p + 8);
  return f.v;
}

// B fragment: 32(K) x 16(N) bf16 from "BT" layout [N][K] (K contiguous, stride ldk).
// Lanes 0-15: N=lane, K = k0+0..15 contiguous; lanes 16-31: K = k0+16..31.
__device__ __forceinline__ bfrag_t load_bfrag(const unsigned short* bt, int ldk,
                                              int n0, int k0, int lane) {
  int col = n0 + (lane & 15);
  int kk  = k0 + ((lane >> 4) << 4);
  const unsigned short* p = bt + (size_t)col * ldk + kk;
  FragU f;
  f.q[0] = *(const U128*)(p);
  f.q[1] = *(const U128*)(p + 8);
  return f.v;
}

// ---------------------------------------------------------------------------
// K0a: fp32 -> bf16 bulk convert (x)
__global__ void f32_to_bf16_kernel(const float* __restrict__ src,
                                   unsigned short* __restrict__ dst, int n) {
  int i = blockIdx.x * blockDim.x + threadIdx.x;
  if (i < n) dst[i] = f2bf(src[i]);
}

// K0b: transpose the four 1024x1024 weights into [N][K] bf16
__global__ void transpose_w_kernel(const float* __restrict__ w0, const float* __restrict__ w1,
                                   const float* __restrict__ w2, const float* __restrict__ w3,
                                   unsigned short* __restrict__ t0, unsigned short* __restrict__ t1,
                                   unsigned short* __restrict__ t2, unsigned short* __restrict__ t3) {
  const float* src; unsigned short* dst;
  switch (blockIdx.y) {
    case 0:  src = w0; dst = t0; break;
    case 1:  src = w1; dst = t1; break;
    case 2:  src = w2; dst = t2; break;
    default: src = w3; dst = t3; break;
  }
  int idx = blockIdx.x * blockDim.x + threadIdx.x;   // over 1024*1024
  int k = idx >> 10, n = idx & 1023;
  dst[(size_t)n * 1024 + k] = f2bf(src[idx]);
}

// K0c: prepend mem_k/mem_v, zero padding rows/cols of Kb and Vt
__global__ void prep_kv_kernel(const float* __restrict__ mem_k, const float* __restrict__ mem_v,
                               unsigned short* __restrict__ Kb, unsigned short* __restrict__ Vt) {
  int p = blockIdx.x;            // plane = b*16 + h  (32 planes)
  int h = p & 15;
  int t = threadIdx.x;
  for (int e = t; e < MEM_ * DH_; e += blockDim.x) {          // Kb mem rows 0..15
    int jm = e >> 6, d = e & 63;
    Kb[((size_t)p * JK_ + jm) * DH_ + d] = f2bf(mem_k[(h * MEM_ + jm) * DH_ + d]);
  }
  for (int e = t; e < (JK_ - J_) * DH_; e += blockDim.x)      // Kb pad rows 2064..2111
    Kb[((size_t)p * JK_ + J_) * DH_ + e] = 0;
  for (int e = t; e < MEM_ * DH_; e += blockDim.x) {          // Vt mem cols j=0..15
    int jm = e >> 6, d = e & 63;
    Vt[((size_t)p * DH_ + d) * JP_ + jm] = f2bf(mem_v[(h * MEM_ + jm) * DH_ + d]);
  }
  for (int e = t; e < DH_ * (JP_ - J_); e += blockDim.x) {    // Vt pad cols 2064..2079
    int d = e >> 4, j = J_ + (e & 15);
    Vt[((size_t)p * DH_ + d) * JP_ + j] = 0;
  }
}

// ---------------------------------------------------------------------------
// K1: QKV projections, bf16 WMMA GEMM (M=4096, N=1024, K=1024), head layouts out.
__device__ __forceinline__ void store_qkv(int mode, unsigned short* Qb, unsigned short* Kb,
                                          unsigned short* Vt, int r, int c, float val) {
  int b = r >> 11, i = r & 2047;
  int h = c >> 6,  d = c & 63;
  unsigned short bv = f2bf(val);
  if (mode == 0)      Qb[((size_t)(b * 16 + h) * N_  + i) * DH_ + d] = bv;
  else if (mode == 1) Kb[((size_t)(b * 16 + h) * JK_ + MEM_ + i) * DH_ + d] = bv;
  else                Vt[((size_t)(b * 16 + h) * DH_ + d) * JP_ + MEM_ + i] = bv;
}

__global__ void qkv_gemm_kernel(const unsigned short* __restrict__ xb,
                                const unsigned short* __restrict__ WqT,
                                const unsigned short* __restrict__ WkT,
                                const unsigned short* __restrict__ WvT,
                                unsigned short* __restrict__ Qb,
                                unsigned short* __restrict__ Kb,
                                unsigned short* __restrict__ Vt) {
  int mode = blockIdx.z;
  const unsigned short* bt = (mode == 0) ? WqT : (mode == 1) ? WkT : WvT;
  int lane = threadIdx.x & 31, wid = threadIdx.x >> 5;
  int M0 = blockIdx.x * 64 + (wid >> 1) * 16;
  int N0 = blockIdx.y * 64 + (wid & 1) * 32;
  v8f acc0 = vzero(), acc1 = vzero();
  for (int k0 = 0; k0 < DIM_; k0 += 32) {
    __builtin_prefetch((const void*)(xb + (size_t)M0 * DIM_ + k0 + 256), 0, 3);
    bfrag_t af = load_afrag(xb, DIM_, M0, k0, lane);
    bfrag_t b0 = load_bfrag(bt, DIM_, N0,      k0, lane);
    bfrag_t b1 = load_bfrag(bt, DIM_, N0 + 16, k0, lane);
    acc0 = wmma_bf16(af, b0, acc0);
    acc1 = wmma_bf16(af, b1, acc1);
  }
  int col   = N0 + (lane & 15);
  int rbase = M0 + ((lane >> 4) << 3);
#pragma unroll
  for (int r = 0; r < 8; ++r) {
    store_qkv(mode, Qb, Kb, Vt, rbase + r, col,      acc0[r]);
    store_qkv(mode, Qb, Kb, Vt, rbase + r, col + 16, acc1[r]);
  }
}

// ---------------------------------------------------------------------------
// K2: scores. WG computes dots[16h][16i][32j] with WMMA, mixes via pre_proj in LDS,
// applies scale + causal mask, writes S[b][k1][i][j] fp32.
__global__ void scores_kernel(const unsigned short* __restrict__ Qb,
                              const unsigned short* __restrict__ Kb,
                              const float* __restrict__ pre,
                              float* __restrict__ S) {
  __shared__ float dotsLDS[16 * 520];   // plane stride 16*32 + 8 pad
  __shared__ float spre[256];
  int tid = threadIdx.x, lane = tid & 31, wid = tid >> 5;
  int b = blockIdx.z, it = blockIdx.y, jc = blockIdx.x;
  spre[tid] = pre[tid];

  v8f acc[2][2];
  acc[0][0] = vzero(); acc[0][1] = vzero(); acc[1][0] = vzero(); acc[1][1] = vzero();
#pragma unroll
  for (int hh = 0; hh < 2; ++hh) {
    int h = wid * 2 + hh;
    const unsigned short* qbase = Qb + ((size_t)(b * 16 + h) * N_ + it * 16) * DH_;
    const unsigned short* kbase = Kb + (size_t)(b * 16 + h) * JK_ * DH_;
#pragma unroll
    for (int k0 = 0; k0 < DH_; k0 += 32) {
      bfrag_t af = load_afrag(qbase, DH_, 0, k0, lane);
#pragma unroll
      for (int js = 0; js < 2; ++js) {
        bfrag_t bv = load_bfrag(kbase, DH_, jc * 32 + js * 16, k0, lane);
        acc[hh][js] = wmma_bf16(af, bv, acc[hh][js]);
      }
    }
  }
#pragma unroll
  for (int hh = 0; hh < 2; ++hh) {
    int h = wid * 2 + hh;
#pragma unroll
    for (int js = 0; js < 2; ++js) {
      int col   = js * 16 + (lane & 15);
      int rbase = (lane >> 4) << 3;
#pragma unroll
      for (int r = 0; r < 8; ++r)
        dotsLDS[h * 520 + (rbase + r) * 32 + col] = acc[hh][js][r];
    }
  }
  __syncthreads();

  const float scale = 0.125f;   // 64^-0.5
  for (int p = tid; p < 512; p += 256) {
    int i = p >> 5, j = p & 31;
    int jglob = jc * 32 + j, iglob = it * 16 + i;
    if (jglob >= J_) continue;
    float dv[16];
#pragma unroll
    for (int h = 0; h < 16; ++h) dv[h] = dotsLDS[h * 520 + p];
    bool masked = (jglob > iglob + MEM_);
#pragma unroll
    for (int k1 = 0; k1 < 16; ++k1) {
      float s;
      if (masked) s = NEGMAX;
      else {
        s = 0.f;
#pragma unroll
        for (int h = 0; h < 16; ++h) s += dv[h] * spre[h * 16 + k1];
        s *= scale;
      }
      S[((size_t)(b * 16 + k1) * N_ + iglob) * J_ + jglob] = s;
    }
  }
}

// ---------------------------------------------------------------------------
// K3: per-row exact top-64 threshold (MSB-first radix select on order-preserving
// float keys) + masked softmax -> bf16 attention weights.
__global__ void topk_softmax_kernel(const float* __restrict__ S,
                                    unsigned short* __restrict__ attnA) {
  __shared__ float    rowv[J_];
  __shared__ unsigned hist[256];
  __shared__ unsigned ssuf[256];
  __shared__ float    red[256];
  __shared__ unsigned s_prefix;
  __shared__ int      s_want;
  int t = threadIdx.x;
  int i = blockIdx.x, k1 = blockIdx.y, b = blockIdx.z;
  const float* srow = S + ((size_t)(b * 16 + k1) * N_ + i) * J_;
  for (int j = t; j < J_; j += 256) rowv[j] = srow[j];
  if (t == 0) { s_prefix = 0u; s_want = TOPK_; }
  __syncthreads();

  for (int pass = 0; pass < 4; ++pass) {
    int shift = 24 - pass * 8;
    unsigned himask = (pass == 0) ? 0u : (0xFFFFFFFFu << (shift + 8));
    hist[t] = 0u;
    __syncthreads();
    unsigned pref = s_prefix;
    for (int j = t; j < J_; j += 256) {
      unsigned u = __float_as_uint(rowv[j]);
      unsigned key = u ^ ((u >> 31) ? 0xFFFFFFFFu : 0x80000000u);
      if ((key & himask) == (pref & himask))
        atomicAdd(&hist[(key >> shift) & 255u], 1u);
    }
    __syncthreads();
    ssuf[t] = hist[t];
    __syncthreads();
    for (int off = 1; off < 256; off <<= 1) {     // suffix-sum (count >= bin)
      unsigned add = (t + off < 256) ? ssuf[t + off] : 0u;
      __syncthreads();
      ssuf[t] += add;
      __syncthreads();
    }
    int wantv = s_want;
    unsigned sufT = ssuf[t];
    unsigned sufN = (t < 255) ? ssuf[t + 1] : 0u;
    if ((int)sufT >= wantv && (t == 255 || (int)sufN < wantv)) {
      s_prefix = s_prefix | ((unsigned)t << shift);
      s_want   = wantv - (int)sufN;
    }
    __syncthreads();
  }
  unsigned thr_key = s_prefix;   // key of 64th largest; keep key >= thr_key (ties kept)

  float lmax = NEGMAX;
  for (int j = t; j < J_; j += 256) lmax = fmaxf(lmax, rowv[j]);
  red[t] = lmax; __syncthreads();
  for (int s = 128; s > 0; s >>= 1) { if (t < s) red[t] = fmaxf(red[t], red[t + s]); __syncthreads(); }
  float gmax = red[0];
  __syncthreads();
  float lsum = 0.f;
  for (int j = t; j < J_; j += 256) {
    unsigned u = __float_as_uint(rowv[j]);
    unsigned key = u ^ ((u >> 31) ? 0xFFFFFFFFu : 0x80000000u);
    if (key >= thr_key) lsum += __expf(rowv[j] - gmax);
  }
  red[t] = lsum; __syncthreads();
  for (int s = 128; s > 0; s >>= 1) { if (t < s) red[t] += red[t + s]; __syncthreads(); }
  float inv = 1.0f / red[0];
  unsigned short* arow = attnA + ((size_t)(b * 16 + k1) * N_ + i) * J_;
  for (int j = t; j < J_; j += 256) {
    unsigned u = __float_as_uint(rowv[j]);
    unsigned key = u ^ ((u >> 31) ? 0xFFFFFFFFu : 0x80000000u);
    float p = (key >= thr_key) ? __expf(rowv[j] - gmax) * inv : 0.f;
    arow[j] = f2bf(p);
  }
}

// ---------------------------------------------------------------------------
// K4: post_proj head mix: attn2[b,k2,i,j] = sum_k1 attn[b,k1,i,j] * post[k1,k2]
__global__ void mix_attn_kernel(const unsigned short* __restrict__ attnA,
                                const float* __restrict__ post,
                                unsigned short* __restrict__ attn2) {
  __shared__ float spost[256];
  int t = threadIdx.x;
  spost[t] = post[t];
  __syncthreads();
  long idx = (long)blockIdx.x * 256 + t;
  const long total = (long)B_ * N_ * JP_;
  if (idx >= total) return;
  int b = (int)(idx / ((long)N_ * JP_));
  int rem = (int)(idx - (long)b * N_ * JP_);
  int i = rem / JP_, j = rem % JP_;
  float a[16];
  if (j < J_) {
#pragma unroll
    for (int k1 = 0; k1 < 16; ++k1)
      a[k1] = bf2f(attnA[((size_t)(b * 16 + k1) * N_ + i) * J_ + j]);
  } else {
#pragma unroll
    for (int k1 = 0; k1 < 16; ++k1) a[k1] = 0.f;   // zero the j padding
  }
#pragma unroll
  for (int k2 = 0; k2 < 16; ++k2) {
    float o = 0.f;
#pragma unroll
    for (int k1 = 0; k1 < 16; ++k1) o += a[k1] * spost[k1 * 16 + k2];
    attn2[((size_t)(b * 16 + k2) * N_ + i) * JP_ + j] = f2bf(o);
  }
}

// ---------------------------------------------------------------------------
// K5: ctx = attn2 @ V via WMMA (K = 2080, V pre-transposed to [d][j]).
__global__ void av_gemm_kernel(const unsigned short* __restrict__ attn2,
                               const unsigned short* __restrict__ Vt,
                               unsigned short* __restrict__ ctx) {
  int lane = threadIdx.x & 31, wid = threadIdx.x >> 5;   // wid = d-tile 0..3
  int b = blockIdx.z, k2 = blockIdx.y, it = blockIdx.x;
  const unsigned short* abase = attn2 + ((size_t)(b * 16 + k2) * N_ + it * 16) * JP_;
  const unsigned short* vbase = Vt + (size_t)(b * 16 + k2) * DH_ * JP_;
  v8f acc = vzero();
  for (int j0 = 0; j0 < JP_; j0 += 32) {
    bfrag_t af = load_afrag(abase, JP_, 0, j0, lane);
    bfrag_t bv = load_bfrag(vbase, JP_, wid * 16, j0, lane);
    acc = wmma_bf16(af, bv, acc);
  }
  int dcol  = wid * 16 + (lane & 15);
  int rbase = (lane >> 4) << 3;
#pragma unroll
  for (int r = 0; r < 8; ++r) {
    int iglob = it * 16 + rbase + r;
    ctx[((size_t)b * N_ + iglob) * DIM_ + k2 * DH_ + dcol] = f2bf(acc[r]);
  }
}

// ---------------------------------------------------------------------------
// K6: out = ctx @ Wo + bo (WMMA, fp32 result).
// CDNA5 async path: the 64x32 bf16 Wo tile shared by all 8 waves is staged in
// LDS with GLOBAL_LOAD_ASYNC_TO_LDS_B128 (ASYNCcnt), double-buffered so the
// next tile's DMA overlaps the current tile's WMMAs.
__global__ void out_gemm_kernel(const unsigned short* __restrict__ ctx,
                                const unsigned short* __restrict__ WoT,
                                const float* __restrict__ bo,
                                float* __restrict__ out) {
  __shared__ unsigned short Bsm[2][64 * 32];   // [buf][col*32 + k], 8 KB total
  int t = threadIdx.x;
  int lane = t & 31, wid = t >> 5;
  int Mblk = blockIdx.x * 64, Nblk = blockIdx.y * 64;
  int M0   = Mblk + (wid >> 1) * 16;
  int nloc = (wid & 1) * 32;                   // wave's local column base in tile

  // staging assignment: 256 threads x 16B = 4 KB tile
  int scol = t >> 2;                           // 0..63 local column
  int skp  = (t & 3) * 8;                      // 0,8,16,24 (bf16 elements)
  const unsigned short* gsrc = WoT + (size_t)(Nblk + scol) * DIM_ + skp;

  auto stage = [&](int buf, int k0) {
#if defined(__gfx1250__)
    unsigned lds_off = (unsigned)(unsigned long long)&Bsm[buf][scol * 32 + skp];
    unsigned long long ga = (unsigned long long)(gsrc + k0);
    asm volatile("global_load_async_to_lds_b128 %0, %1, off"
                 :: "v"(lds_off), "v"(ga) : "memory");
#else
    *(U128*)((unsigned short*)&Bsm[buf][scol * 32 + skp]) = *(const U128*)(gsrc + k0);
#endif
  };

  v8f acc0 = vzero(), acc1 = vzero();
  stage(0, 0);
  const int KSTEPS = DIM_ / 32;                // 32
  for (int ks = 0; ks < KSTEPS; ++ks) {
    if (ks + 1 < KSTEPS) {
      stage((ks + 1) & 1, (ks + 1) * 32);      // prefetch next tile (in flight)
#if defined(__gfx1250__)
      asm volatile("s_wait_asynccnt 0x1" ::: "memory");  // current tile done
#endif
    } else {
#if defined(__gfx1250__)
      asm volatile("s_wait_asynccnt 0x0" ::: "memory");  // last tile done
#endif
    }
    __syncthreads();
    const unsigned short* bsm = &Bsm[ks & 1][0];
    bfrag_t af = load_afrag(ctx, DIM_, M0, ks * 32, lane);
    bfrag_t b0 = load_bfrag(bsm, 32, nloc,      0, lane);
    bfrag_t b1 = load_bfrag(bsm, 32, nloc + 16, 0, lane);
    acc0 = wmma_bf16(af, b0, acc0);
    acc1 = wmma_bf16(af, b1, acc1);
    __syncthreads();                           // protect buffer before re-fill
  }

  int col   = Nblk + nloc + (lane & 15);
  int rbase = M0 + ((lane >> 4) << 3);
#pragma unroll
  for (int r = 0; r < 8; ++r) {
    out[(size_t)(rbase + r) * DIM_ + col]      = acc0[r] + bo[col];
    out[(size_t)(rbase + r) * DIM_ + col + 16] = acc1[r] + bo[col + 16];
  }
}

// ---------------------------------------------------------------------------
extern "C" void kernel_launch(void* const* d_in, const int* in_sizes, int n_in,
                              void* d_out, int out_size, void* d_ws, size_t ws_size,
                              hipStream_t stream) {
  const float* x    = (const float*)d_in[0];
  const float* Wq   = (const float*)d_in[1];
  const float* Wk   = (const float*)d_in[2];
  const float* Wv   = (const float*)d_in[3];
  const float* pre  = (const float*)d_in[4];
  const float* post = (const float*)d_in[5];
  const float* memk = (const float*)d_in[6];
  const float* memv = (const float*)d_in[7];
  const float* Wo   = (const float*)d_in[8];
  const float* bo   = (const float*)d_in[9];
  float* out = (float*)d_out;
  (void)in_sizes; (void)n_in; (void)out_size; (void)ws_size;

  char* wsb = (char*)d_ws;
  size_t off = 0;
  auto take = [&](size_t bytes) {
    char* p = wsb + off;
    off = (off + bytes + 255) & ~(size_t)255;
    return p;
  };
  unsigned short* xb    = (unsigned short*)take((size_t)B_ * N_ * DIM_ * 2);
  unsigned short* WqT   = (unsigned short*)take((size_t)DIM_ * DIM_ * 2);
  unsigned short* WkT   = (unsigned short*)take((size_t)DIM_ * DIM_ * 2);
  unsigned short* WvT   = (unsigned short*)take((size_t)DIM_ * DIM_ * 2);
  unsigned short* WoT   = (unsigned short*)take((size_t)DIM_ * DIM_ * 2);
  unsigned short* Qb    = (unsigned short*)take((size_t)B_ * H_ * N_ * DH_ * 2);
  unsigned short* Kb    = (unsigned short*)take((size_t)B_ * H_ * JK_ * DH_ * 2);
  unsigned short* Vt    = (unsigned short*)take((size_t)B_ * H_ * DH_ * JP_ * 2);
  float*          S     = (float*)take((size_t)B_ * H_ * N_ * J_ * 4);
  unsigned short* attnA = (unsigned short*)take((size_t)B_ * H_ * N_ * J_ * 2);
  unsigned short* attn2 = (unsigned short*)take((size_t)B_ * H_ * N_ * JP_ * 2);
  unsigned short* ctx   = (unsigned short*)take((size_t)B_ * N_ * DIM_ * 2);

  f32_to_bf16_kernel<<<(B_ * N_ * DIM_ + 255) / 256, 256, 0, stream>>>(x, xb, B_ * N_ * DIM_);
  transpose_w_kernel<<<dim3(4096, 4), 256, 0, stream>>>(Wq, Wk, Wv, Wo, WqT, WkT, WvT, WoT);
  prep_kv_kernel<<<32, 256, 0, stream>>>(memk, memv, Kb, Vt);
  qkv_gemm_kernel<<<dim3(64, 16, 3), 256, 0, stream>>>(xb, WqT, WkT, WvT, Qb, Kb, Vt);
  scores_kernel<<<dim3(66, 128, 2), 256, 0, stream>>>(Qb, Kb, pre, S);
  topk_softmax_kernel<<<dim3(2048, 16, 2), 256, 0, stream>>>(S, attnA);
  mix_attn_kernel<<<(int)(((long)B_ * N_ * JP_ + 255) / 256), 256, 0, stream>>>(attnA, post, attn2);
  av_gemm_kernel<<<dim3(128, 16, 2), 128, 0, stream>>>(attn2, Vt, ctx);
  out_gemm_kernel<<<dim3(64, 16), 256, 0, stream>>>(ctx, WoT, bo, out);
}
